// CRFModel_72722386256260
// MI455X (gfx1250) — matrix operand order repaired and checked by reading, not compiled, not run on hardware
//
#include <hip/hip_runtime.h>

#define DEV __device__ __forceinline__

typedef __attribute__((ext_vector_type(16))) __bf16 v16bf;
typedef __attribute__((ext_vector_type(8)))  __bf16 v8bf;
typedef __attribute__((ext_vector_type(8)))  float  v8f;

// Problem dims
#define BB   32
#define TT   128
#define HH   256
#define EE   300
#define LEXD 399
#define CTXD 20
#define DD   699
#define DP   704      // D padded to mult of 32
#define G3   768      // 3*H
#define KK   600      // tags
#define KKP  608      // tags padded to mult of 16
#define CCD  532      // 2H + CTX
#define CCP  544      // padded to mult of 32
#define MM   4096     // B*T
#define NEGV (-10000.0f)

// ---------- helpers ----------
DEV __bf16 f2bf(float f) {
  unsigned u = __builtin_bit_cast(unsigned, f);
  unsigned r = u + 0x7FFFu + ((u >> 16) & 1u);   // round-to-nearest-even
  unsigned short h = (unsigned short)(r >> 16);
  return __builtin_bit_cast(__bf16, h);
}

DEV float sigmoidf(float x) { return 1.0f / (1.0f + __expf(-x)); }

DEV float tanh_fast(float x) {
  x = fminf(fmaxf(x, -15.0f), 15.0f);
  float e = __expf(-2.0f * x);
  return (1.0f - e) / (1.0f + e);
}

// Register-blocked WMMA: MB x NB grid of 16x16 C-tiles of C = A * W^T.
// A row-major [*, lda] bf16 (global or LDS), W row-major [N, ldb] bf16.
// A-fragments are loaded once per k-step and reused across NB columns;
// B-fragments reused across MB rows -> (MB*2 + NB*2) b128 loads per MB*NB WMMAs.
template<int MB, int NB>
DEV void wmma_block(const __bf16* A, int lda, const __bf16* W, int ldb,
                    int m0, int n0, int lane, int ksteps, v8f acc[MB][NB]) {
  const int half = lane >> 4;
  const int r    = lane & 15;
  for (int ks = 0; ks < ksteps; ++ks) {
    const int kb = ks * 32;
    v16bf a[MB];
#pragma unroll
    for (int mb = 0; mb < MB; ++mb) {
      const __bf16* arow = A + (size_t)(m0 + mb * 16 + r) * lda;
      v8bf alo = *(const v8bf*)(arow + kb + half * 16);
      v8bf ahi = *(const v8bf*)(arow + kb + half * 16 + 8);
      a[mb] = __builtin_shufflevector(alo, ahi,
                                      0,1,2,3,4,5,6,7,8,9,10,11,12,13,14,15);
    }
#pragma unroll
    for (int nb = 0; nb < NB; ++nb) {
      const __bf16* brow = W + (size_t)(n0 + nb * 16 + r) * ldb + half * 16;
      v16bf b = *(const v16bf*)(brow + kb);
#pragma unroll
      for (int mb = 0; mb < MB; ++mb)
        acc[mb][nb] = __builtin_amdgcn_wmma_f32_16x16x32_bf16(
            false, a[mb], false, b, (short)0, acc[mb][nb], false, false);
    }
  }
}

// C/D layout: VGPR j -> M = m0 + j + 8*(lane>=16), N = n0 + (lane&15)
DEV void store_tile(float* C, int ldc, int m0, int n0, int lane, int nmax,
                    const float* bias, const v8f& acc) {
  const int half = lane >> 4;
  const int n = n0 + (lane & 15);
  if (n >= nmax) return;
  const float bv = bias ? bias[n] : 0.0f;
#pragma unroll
  for (int j = 0; j < 8; ++j)
    C[(size_t)(m0 + half * 8 + j) * ldc + n] = acc[j] + bv;
}

// ---------- kernels ----------

// f32 [srows,scols] -> bf16 [drows,dcols] with zero padding.
__global__ void cvt_pad_kernel(const float* __restrict__ src, __bf16* __restrict__ dst,
                               int srows, int scols, int dcols) {
  const int row = blockIdx.x;
  const bool rv = row < srows;
  for (int c = threadIdx.x; c < dcols; c += blockDim.x) {
    float v = (rv && c < scols) ? src[(size_t)row * scols + c] : 0.0f;
    dst[(size_t)row * dcols + c] = f2bf(v);
  }
}

// Build inp_bf16 [MM, DP] = [emb | x_lex | 0], and fill ctx+pad tail of concat buffer.
__global__ void embed_concat_kernel(const int* __restrict__ xemb,
                                    const float* __restrict__ xlex,
                                    const float* __restrict__ xctx,
                                    const float* __restrict__ etab,
                                    __bf16* __restrict__ inp,
                                    __bf16* __restrict__ concat) {
  const int m = blockIdx.x;
  const int idx = xemb[m];
  for (int c = threadIdx.x; c < DP; c += blockDim.x) {
    float v;
    if (c < EE)       v = etab[(size_t)idx * EE + c];
    else if (c < DD)  v = xlex[(size_t)m * LEXD + (c - EE)];
    else              v = 0.0f;
    inp[(size_t)m * DP + c] = f2bf(v);
  }
  for (int c = threadIdx.x; c < (CCP - 2 * HH); c += blockDim.x) {
    float v = (c < CTXD) ? xctx[(size_t)m * CTXD + c] : 0.0f;
    concat[(size_t)m * CCP + 2 * HH + c] = f2bf(v);
  }
}

// Register-blocked wave-per-supertile GEMM: C = A * W^T + bias
template<int MB, int NB>
__global__ void gemm_blk_kernel(const __bf16* __restrict__ A, const __bf16* __restrict__ W,
                                float* __restrict__ C, const float* __restrict__ bias,
                                int lda, int ldb, int ldc,
                                int msup, int nsup, int nmax, int ksteps) {
  const int st   = blockIdx.x * (blockDim.x >> 5) + (threadIdx.x >> 5);
  const int lane = threadIdx.x & 31;
  if (st >= msup * nsup) return;   // wave-uniform
  const int ms = st % msup, ns = st / msup;
  const int m0 = ms * (16 * MB), n0 = ns * (16 * NB);
  v8f acc[MB][NB] = {};
  wmma_block<MB, NB>(A, lda, W, ldb, m0, n0, lane, ksteps, acc);
#pragma unroll
  for (int mb = 0; mb < MB; ++mb)
#pragma unroll
    for (int nb = 0; nb < NB; ++nb)
      store_tile(C, ldc, m0 + mb * 16, n0 + nb * 16, lane, nmax, bias, acc[mb][nb]);
}

// Persistent GRU: one workgroup per direction, h resident in LDS.
// Per step: 32x768 = 2x48 C-tiles; wave w owns (mt = w&1, nt = 3*(w>>1)..+2),
// so each LDS A-fragment is loaded once per k-step and reused across 3 WMMAs.
// An opaque per-step SGPR offset keeps the Whh B-fragment loads inside the time
// loop (L2-resident streaming) instead of hoisting+spilling 192 VGPRs to scratch,
// while preserving the kernarg pointer base so loads stay global_load (not flat).
__global__ void __launch_bounds__(1024)
gru_kernel(const float* __restrict__ xp_f, const float* __restrict__ xp_b,
           const __bf16* __restrict__ WhhF, const __bf16* __restrict__ WhhB,
           const float* __restrict__ bhhF, const float* __restrict__ bhhB,
           const float* __restrict__ h0, __bf16* __restrict__ concat) {
  const int dir = blockIdx.x;
  const float*  xp  = dir ? xp_b : xp_f;
  const __bf16* Whh = dir ? WhhB : WhhF;
  const float*  bhh = dir ? bhhB : bhhF;

  __shared__ float  h32[BB * HH];   // 32 KB
  __shared__ __bf16 hbf[BB * HH];   // 16 KB
  __shared__ float  gh[BB * G3];    // 96 KB

  const int tid = threadIdx.x;
  for (int e = tid; e < BB * HH; e += 1024) {
    float v = h0[dir * BB * HH + e];
    h32[e] = v; hbf[e] = f2bf(v);
  }
  __syncthreads();

  const int wave = tid >> 5, lane = tid & 31;
  const int mt  = wave & 1;
  const int nt0 = (wave >> 1) * 3;
  for (int s = 0; s < TT; ++s) {
    const int t = dir ? (TT - 1 - s) : s;   // backward dir consumes reversed time
    // gh = h @ Whh^T
    {
      size_t woff = 0;
      asm volatile("" : "+s"(woff));   // opaque offset: defeat invariant hoisting
      v8f acc[1][3] = {};
      wmma_block<1, 3>(hbf, HH, Whh + woff, HH, mt * 16, nt0 * 16, lane, HH / 32, acc);
#pragma unroll
      for (int nb = 0; nb < 3; ++nb)
        store_tile(gh, G3, mt * 16, (nt0 + nb) * 16, lane, G3, nullptr, acc[0][nb]);
    }
    __syncthreads();
    // gates + state update
    for (int e = tid; e < BB * HH; e += 1024) {
      const int b = e >> 8, j = e & 255;
      const float* xrow = xp + (size_t)(b * TT + t) * G3;
      float hr = gh[b * G3 + j]          + bhh[j];
      float hz = gh[b * G3 + HH + j]     + bhh[HH + j];
      float hn = gh[b * G3 + 2 * HH + j] + bhh[2 * HH + j];
      float r = sigmoidf(xrow[j] + hr);
      float z = sigmoidf(xrow[HH + j] + hz);
      float n = tanh_fast(xrow[2 * HH + j] + r * hn);
      float hv = (1.0f - z) * n + z * h32[e];
      h32[e] = hv; hbf[e] = f2bf(hv);
      concat[(size_t)(b * TT + t) * CCP + dir * HH + j] = f2bf(hv);
    }
    __syncthreads();
  }
}

// Batched Viterbi forward pass: block per batch element, thread per next-tag.
__global__ void __launch_bounds__(640)
viterbi_kernel(const float* __restrict__ feats, const float* __restrict__ trans,
               unsigned short* __restrict__ bptr, float* __restrict__ fv_final) {
  const int b = blockIdx.x, tid = threadIdx.x;
  __shared__ float fv[KK];
  __shared__ float fvn[KK];
  if (tid < KK) fv[tid] = (tid == KK - 2) ? 0.0f : NEGV;  // START = K-2
  __syncthreads();
  const float* trow = trans + (size_t)tid * KK;           // trans[next, prev]
  for (int t = 0; t < TT; ++t) {
    if (tid < KK) {
      if (t + 1 < TT)
        __builtin_prefetch(&feats[((size_t)b * TT + t + 1) * KK + tid], 0, 1);
      float best = -3.4e38f; int arg = 0;
      for (int p = 0; p < KK; ++p) {
        float sc = fv[p] + trow[p];
        if (sc > best) { best = sc; arg = p; }            // first-max tie-break
      }
      fvn[tid] = best + feats[((size_t)b * TT + t) * KK + tid];
      bptr[((size_t)t * BB + b) * KK + tid] = (unsigned short)arg;
    }
    __syncthreads();
    if (tid < KK) fv[tid] = fvn[tid];
    __syncthreads();
  }
  if (tid < KK) fv_final[(size_t)b * KK + tid] = fv[tid];
}

// Terminal argmax + backtrace; scores -> out[0:32], path (as float) -> out[32:]
__global__ void backtrace_kernel(const float* __restrict__ fv_final,
                                 const float* __restrict__ trans,
                                 const unsigned short* __restrict__ bptr,
                                 float* __restrict__ out) {
  const int b = threadIdx.x;
  if (b >= BB) return;
  const float* trow = trans + (size_t)(KK - 1) * KK;      // STOP = K-1
  float best = -3.4e38f; int arg = 0;
  for (int k = 0; k < KK; ++k) {
    float sc = fv_final[(size_t)b * KK + k] + trow[k];
    if (sc > best) { best = sc; arg = k; }
  }
  out[b] = best;
  float* path = out + BB + (size_t)b * TT;
  int tag = arg;
  path[TT - 1] = (float)tag;
  for (int t = TT - 1; t >= 1; --t) {
    tag = bptr[((size_t)t * BB + b) * KK + tag];
    path[t - 1] = (float)tag;
  }
}

// ---------- launcher ----------
extern "C" void kernel_launch(void* const* d_in, const int* in_sizes, int n_in,
                              void* d_out, int out_size, void* d_ws, size_t ws_size,
                              hipStream_t stream) {
  const float* x_lex = (const float*)d_in[0];
  const int*   x_emb = (const int*)d_in[1];
  const float* x_ctx = (const float*)d_in[2];
  const float* h0    = (const float*)d_in[3];
  const float* etab  = (const float*)d_in[4];
  const float* Wih_f = (const float*)d_in[5];
  const float* Whh_f = (const float*)d_in[6];
  const float* bih_f = (const float*)d_in[7];
  const float* bhh_f = (const float*)d_in[8];
  const float* Wih_b = (const float*)d_in[9];
  const float* Whh_b = (const float*)d_in[10];
  const float* bih_b = (const float*)d_in[11];
  const float* bhh_b = (const float*)d_in[12];
  const float* W_out = (const float*)d_in[13];
  const float* b_out = (const float*)d_in[14];
  const float* trans = (const float*)d_in[15];
  float* out = (float*)d_out;

  char* p = (char*)d_ws;
  auto take = [&](size_t bytes) -> void* {
    void* r = (void*)p;
    p += (bytes + 255) & ~(size_t)255;
    return r;
  };
  __bf16* inp_bf   = (__bf16*)take((size_t)MM * DP * 2);
  __bf16* wihf_bf  = (__bf16*)take((size_t)G3 * DP * 2);
  __bf16* wihb_bf  = (__bf16*)take((size_t)G3 * DP * 2);
  __bf16* whhf_bf  = (__bf16*)take((size_t)G3 * HH * 2);
  __bf16* whhb_bf  = (__bf16*)take((size_t)G3 * HH * 2);
  __bf16* wout_bf  = (__bf16*)take((size_t)KKP * CCP * 2);
  float*  xp_f     = (float*)take((size_t)MM * G3 * 4);
  float*  xp_b     = (float*)take((size_t)MM * G3 * 4);
  __bf16* concat   = (__bf16*)take((size_t)MM * CCP * 2);
  float*  feats    = (float*)take((size_t)MM * KK * 4);
  unsigned short* bptr = (unsigned short*)take((size_t)TT * BB * KK * 2);
  float*  fv_final = (float*)take((size_t)BB * KK * 4);
  (void)ws_size; (void)in_sizes; (void)n_in; (void)out_size;

  // 1) weight conversion (f32 -> bf16, zero-padded)
  cvt_pad_kernel<<<G3,  256, 0, stream>>>(Wih_f, wihf_bf, G3, DD, DP);
  cvt_pad_kernel<<<G3,  256, 0, stream>>>(Wih_b, wihb_bf, G3, DD, DP);
  cvt_pad_kernel<<<G3,  256, 0, stream>>>(Whh_f, whhf_bf, G3, HH, HH);
  cvt_pad_kernel<<<G3,  256, 0, stream>>>(Whh_b, whhb_bf, G3, HH, HH);
  cvt_pad_kernel<<<KKP, 256, 0, stream>>>(W_out, wout_bf, KK, CCD, CCP);

  // 2) embedding gather + concat assembly
  embed_concat_kernel<<<MM, 256, 0, stream>>>(x_emb, x_lex, x_ctx, etab, inp_bf, concat);

  // 3) input projections: xp = inp @ Wih^T + bih (M=4096, N=768, K=704)
  //    2x4 super-tiles: 128 x 12 = 1536 waves -> 192 blocks of 8 waves
  gemm_blk_kernel<2, 4><<<192, 256, 0, stream>>>(inp_bf, wihf_bf, xp_f, bih_f,
                                                 DP, DP, G3, 128, 12, G3, DP / 32);
  gemm_blk_kernel<2, 4><<<192, 256, 0, stream>>>(inp_bf, wihb_bf, xp_b, bih_b,
                                                 DP, DP, G3, 128, 12, G3, DP / 32);

  // 4) bidirectional GRU recurrence (persistent, LDS-resident state)
  gru_kernel<<<2, 1024, 0, stream>>>(xp_f, xp_b, whhf_bf, whhb_bf, bhh_f, bhh_b,
                                     h0, concat);

  // 5) output projection: feats = concat @ W_out^T + b_out (M=4096, N=608, K=544)
  //    2x2 super-tiles: 128 x 19 = 2432 waves -> 304 blocks of 8 waves
  gemm_blk_kernel<2, 2><<<304, 256, 0, stream>>>(concat, wout_bf, feats, b_out,
                                                 CCP, CCP, KK, 128, 19, KK, CCP / 32);

  // 6) Viterbi forward + 7) backtrace
  viterbi_kernel<<<BB, 640, 0, stream>>>(feats, trans, bptr, fv_final);
  backtrace_kernel<<<1, 32, 0, stream>>>(fv_final, trans, bptr, out);
}
